// ScaleAdaptiveDecoderLayer_39367670235433
// MI455X (gfx1250) — matrix operand order repaired and checked by reading, not compiled, not run on hardware
//
#include <hip/hip_runtime.h>

// ---------------------------------------------------------------------------
// ScaleAdaptiveDecoderLayer on gfx1250 (MI455X).
//
// Precision: f32 -> f16 activations/weights, f32 accumulation via
// v_wmma_f32_16x16x32_f16. GEMMs stage the A strip into LDS with the CDNA5
// async-copy path (global_load_async_to_lds_b128 + s_wait_asynccnt), and the
// B fragments are software-pipelined so WMMA overlaps with VMEM.
// ---------------------------------------------------------------------------

#define D_MODEL 256
#define N_HEADS 8
#define HEAD_DIM 32
#define D_FFN 1024
#define BATCH 4
#define SEQ 1024
#define ROWS (BATCH * SEQ) // 4096

typedef _Float16 f16;
typedef _Float16 v8h  __attribute__((ext_vector_type(8)));
typedef _Float16 v16h __attribute__((ext_vector_type(16)));
typedef float    v8f  __attribute__((ext_vector_type(8)));

static __device__ __forceinline__ v8f wmma_f32_f16(v16h a, v16h b, v8f c) {
  // (neg_a, A, neg_b, B, c_mod, C, reuse_a, reuse_b)
  return __builtin_amdgcn_wmma_f32_16x16x32_f16(false, a, false, b, (short)0, c,
                                                false, false);
}

// ISA aperture rule (cdna5_isa 10.2): for a generic pointer into LDS, the low
// 32 bits are the wave-relative LDS byte address.
static __device__ __forceinline__ unsigned lds_addr32(const void* p) {
  return (unsigned)(unsigned long long)p;
}

// Async copy 16 bytes/lane from global to LDS (ASYNCcnt-tracked).
static __device__ __forceinline__ void async_g2l_b128(unsigned lds_byte,
                                                      const void* gptr) {
  asm volatile("global_load_async_to_lds_b128 %0, %1, off"
               :: "v"(lds_byte), "v"((unsigned long long)(size_t)gptr)
               : "memory");
}
static __device__ __forceinline__ void wait_async0() {
  asm volatile("s_wait_asynccnt 0x0" ::: "memory");
}

// A fragment: 16(M) x 32(K). baseRow = row 0 of the 16-row tile, row-major,
// leading dim ldk halves. Lane m = lane&15; elems 0..7 = K kb..kb+7,
// elems 8..15 = K kb+16..kb+23, kb = lane<16 ? 0 : 8.   (ISA 7.12.2)
static __device__ __forceinline__ v16h load_a16x32(const f16* __restrict__ baseRow,
                                                   int ldk, int k0) {
  const int lane = threadIdx.x & 31;
  const int m = lane & 15;
  const int kb = (lane < 16) ? 0 : 8;
  const f16* p = baseRow + (size_t)m * ldk + k0 + kb;
  v8h lo = *(const v8h*)(p);
  v8h hi = *(const v8h*)(p + 16);
  v16h r;
#pragma unroll
  for (int i = 0; i < 8; ++i) { r[i] = lo[i]; r[i + 8] = hi[i]; }
  return r;
}

// B fragment: 32(K) x 16(N), source row-major N x K (math-B column n is
// contiguous in K). Lane n = lane&15; elems 0..7 = K kb..kb+7,
// elems 8..15 = K kb+16..kb+23, kb = lane<16 ? 0 : 8.
static __device__ __forceinline__ v16h load_b32x16(const f16* __restrict__ baseRow,
                                                   int ldk, int k0) {
  const int lane = threadIdx.x & 31;
  const int n = lane & 15;
  const int kb = (lane < 16) ? 0 : 8;
  const f16* p = baseRow + (size_t)n * ldk + k0 + kb;
  v8h lo = *(const v8h*)(p);
  v8h hi = *(const v8h*)(p + 16);
  v16h r;
#pragma unroll
  for (int i = 0; i < 8; ++i) { r[i] = lo[i]; r[i + 8] = hi[i]; }
  return r;
}

// ---------------------------------------------------------------------------
// Elementwise prep kernels
// ---------------------------------------------------------------------------
__global__ void cvt_qk_kernel(const float* __restrict__ e,
                              const float* __restrict__ qp,
                              f16* __restrict__ qkh, f16* __restrict__ eh, int n) {
  int i = blockIdx.x * blockDim.x + threadIdx.x;
  if (i >= n) return;
  float ev = e[i];
  eh[i] = (f16)ev;
  qkh[i] = (f16)(ev + qp[i]);
}

__global__ void cvt_f16_kernel(const float* __restrict__ src,
                               f16* __restrict__ dst, int n) {
  int i = blockIdx.x * blockDim.x + threadIdx.x;
  if (i < n) dst[i] = (f16)src[i];
}

// tau[b,h,n] = dot(embed[b,n,:], tau_w[h,:]) + tau_b[h]
__global__ void tau_kernel(const float* __restrict__ embed,
                           const float* __restrict__ tw,
                           const float* __restrict__ tb,
                           float* __restrict__ tau) {
  int idx = blockIdx.x * blockDim.x + threadIdx.x; // B*N*H = 32768
  if (idx >= BATCH * SEQ * N_HEADS) return;
  int h = idx & 7;
  int n = (idx >> 3) & (SEQ - 1);
  int b = idx >> 13;
  const float* e = embed + ((size_t)(b * SEQ + n)) * D_MODEL;
  const float* w = tw + (size_t)h * D_MODEL;
  float s = 0.f;
#pragma unroll 4
  for (int k = 0; k < D_MODEL; ++k) s += e[k] * w[k];
  tau[((size_t)(b * N_HEADS + h)) * SEQ + n] = s + tb[h];
}

// vt[b,h,d,seq] = v[b,seq,h,d]
__global__ void transpose_v_kernel(const f16* __restrict__ v, f16* __restrict__ vt) {
  int i = blockIdx.x * blockDim.x + threadIdx.x; // 1M
  if (i >= BATCH * N_HEADS * HEAD_DIM * SEQ) return;
  int n = i & (SEQ - 1);
  int d = (i >> 10) & 31;
  int h = (i >> 15) & 7;
  int b = i >> 18;
  vt[i] = v[((size_t)(b * SEQ + n)) * D_MODEL + h * HEAD_DIM + d];
}

// ---------------------------------------------------------------------------
// WMMA GEMM:  C[m,n] = sum_k A[m,k] * W[n,k] + bias[n]
// A: (ROWS x K) f16 row-major, staged into LDS via async copies and shared by
// the waves of a block. W: (N x K) f16 row-major. 16x64 tile / wave, B frags
// software-pipelined one k-step ahead.
// ---------------------------------------------------------------------------
template <int K, int NTILES, int WAVES>
__global__ __launch_bounds__(WAVES * 32) void gemm_wmma_kernel(
    const f16* __restrict__ A, const f16* __restrict__ W,
    const float* __restrict__ bias, float* __restrict__ outf,
    f16* __restrict__ outh, int relu) {
  constexpr int N = NTILES * 64;
  constexpr int TPB = (WAVES + NTILES - 1) / NTILES; // m-tiles per block
  __shared__ f16 sA[TPB * 16 * K];

  const int wlocal = threadIdx.x >> 5;
  const int wave = blockIdx.x * WAVES + wlocal;
  const int mt = wave / NTILES;
  const int nt = wave - mt * NTILES;
  const int mt0 = (blockIdx.x * WAVES) / NTILES; // first m-tile of this block
  const int row0 = mt * 16;
  const int n0 = nt * 64;
  const int lane = threadIdx.x & 31;
  const int hi = lane >> 4;
  const int nl = lane & 15;

  // ---- stage the block's A strip (TPB*16 rows x K) into LDS, async -------
  {
    constexpr int CHUNKS = TPB * 16 * K / 8; // 16-byte chunks
    const f16* strip = A + (size_t)mt0 * 16 * K;
    const unsigned lbase = lds_addr32(&sA[0]);
    for (int c = threadIdx.x; c < CHUNKS; c += WAVES * 32)
      async_g2l_b128(lbase + (unsigned)c * 16u, strip + (size_t)c * 8);
    wait_async0();
    __syncthreads();
  }

  const f16* atile = sA + (size_t)(mt - mt0) * 16 * K;

  // ---- pipelined WMMA main loop ------------------------------------------
  v8f acc[4] = {};
  v16h bcur[4];
#pragma unroll
  for (int t = 0; t < 4; ++t)
    bcur[t] = load_b32x16(W + (size_t)(n0 + t * 16) * K, K, 0);

  for (int k0 = 0; k0 < K; k0 += 32) {
    v16h a = load_a16x32(atile, K, k0); // ds_load_b128 x2
    const int kn = (k0 + 32 < K) ? (k0 + 32) : 0; // last iter: dummy prefetch
    v16h bn[4];
#pragma unroll
    for (int t = 0; t < 4; ++t)
      bn[t] = load_b32x16(W + (size_t)(n0 + t * 16) * K, K, kn);
#pragma unroll
    for (int t = 0; t < 4; ++t)
      acc[t] = wmma_f32_f16(a, bcur[t], acc[t]);
#pragma unroll
    for (int t = 0; t < 4; ++t) bcur[t] = bn[t];
  }

  // ---- epilogue: bias (+ReLU), store f32 or f16 --------------------------
#pragma unroll
  for (int t = 0; t < 4; ++t) {
    int col = n0 + t * 16 + nl;
    float bv = bias ? bias[col] : 0.f;
#pragma unroll
    for (int r = 0; r < 8; ++r) {
      int row = row0 + r + 8 * hi; // C layout: VGPR r -> M = r + 8*(lane/16)
      float val = acc[t][r] + bv;
      if (relu) val = val > 0.f ? val : 0.f;
      if (outf) outf[(size_t)row * N + col] = val;
      else      outh[(size_t)row * N + col] = (f16)val;
    }
  }
}

// ---------------------------------------------------------------------------
// Attention: one workgroup (8 waves) per (b, h, 16-query tile).
// LDS: 16 x 1024 f32 score buffer (64 KB), reused for ctx partial reduction.
// ---------------------------------------------------------------------------
__global__ __launch_bounds__(256) void attn_kernel(
    const f16* __restrict__ qh, const f16* __restrict__ kh,
    const f16* __restrict__ vt, const float* __restrict__ tau,
    const float* __restrict__ dist, f16* __restrict__ ctx) {
  __shared__ float sS[16 * SEQ]; // 64 KB

  const int bid = blockIdx.x;      // B * H * (SEQ/16) = 2048
  const int it = bid & 63;
  const int h = (bid >> 6) & 7;
  const int b = bid >> 9;
  const int i0 = it * 16;
  const int tid = threadIdx.x;
  const int wave = tid >> 5;
  const int lane = tid & 31;
  const int hi = lane >> 4;
  const int nl = lane & 15;

  float rtau[8];
  {
    const float* taup = tau + ((size_t)(b * N_HEADS + h)) * SEQ + i0 + 8 * hi;
#pragma unroll
    for (int r = 0; r < 8; ++r) rtau[r] = taup[r];
  }

  const f16* qrow = qh + ((size_t)(b * SEQ + i0)) * D_MODEL + h * HEAD_DIM;
  const v16h aq = load_a16x32(qrow, D_MODEL, 0);

  const float invsq = 0.17677669529663689f; // 1/sqrt(32)
  const float* distb = dist + (size_t)b * SEQ * SEQ;

  // ---- pass 1: scores = QK^T/sqrt(d) + dist*tau -> LDS -------------------
  for (int jt = wave; jt < SEQ / 16; jt += 8) {
    const int j0 = jt * 16;
    const f16* krow = kh + ((size_t)(b * SEQ + j0)) * D_MODEL + h * HEAD_DIM;
    v16h bk = load_b32x16(krow, D_MODEL, 0); // B[k][n] = K[j0+n][k]
    v8f s = {};
    s = wmma_f32_f16(aq, bk, s);
#pragma unroll
    for (int r = 0; r < 8; ++r) {
      int row = r + 8 * hi;
      int col = j0 + nl;
      float v = s[r] * invsq + distb[(size_t)(i0 + row) * SEQ + col] * rtau[r];
      sS[row * SEQ + col] = v;
    }
  }
  __syncthreads();

  // ---- softmax over each row (16 threads / row, shfl_xor reductions) -----
  {
    const int row = tid >> 4;
    const int seg = tid & 15;
    float* rp = sS + row * SEQ + seg * 64;
    float mx = -3.0e38f;
    for (int c = 0; c < 64; ++c) mx = fmaxf(mx, rp[c]);
#pragma unroll
    for (int m = 1; m < 16; m <<= 1) mx = fmaxf(mx, __shfl_xor(mx, m, 32));
    float sum = 0.f;
    for (int c = 0; c < 64; ++c) sum += __expf(rp[c] - mx);
#pragma unroll
    for (int m = 1; m < 16; m <<= 1) sum += __shfl_xor(sum, m, 32);
    float inv = 1.f / sum;
    for (int c = 0; c < 64; ++c) rp[c] = __expf(rp[c] - mx) * inv;
  }
  __syncthreads();

  // ---- pass 2: ctx partial = attn[:, wave*128 : +128] @ V ----------------
  v8f acc0 = {}, acc1 = {};
  const int jbase = wave * 128;
  const int abase = hi ? 8 : 0;
  const int m = nl;
  const f16* vhead = vt + ((size_t)((b * N_HEADS + h) * HEAD_DIM)) * SEQ;
  for (int js = 0; js < 128; js += 32) {
    const int j0 = jbase + js;
    v16h a;
#pragma unroll
    for (int e = 0; e < 8; ++e) { // attn A-frag from LDS, cvt f32->f16
      a[e]     = (f16)sS[m * SEQ + j0 + abase + e];
      a[e + 8] = (f16)sS[m * SEQ + j0 + abase + 16 + e];
    }
    v16h b0 = load_b32x16(vhead + j0, SEQ, 0);                    // dims 0..15
    v16h b1 = load_b32x16(vhead + (size_t)16 * SEQ + j0, SEQ, 0); // dims 16..31
    acc0 = wmma_f32_f16(a, b0, acc0);
    acc1 = wmma_f32_f16(a, b1, acc1);
  }
  __syncthreads(); // all waves done reading sS

  {
    float* pp = sS + wave * 512;
#pragma unroll
    for (int r = 0; r < 8; ++r) {
      pp[(r + 8 * hi) * 32 + nl]      = acc0[r];
      pp[(r + 8 * hi) * 32 + nl + 16] = acc1[r];
    }
  }
  __syncthreads();
  for (int idx = tid; idx < 512; idx += 256) {
    float s = 0.f;
#pragma unroll
    for (int w = 0; w < 8; ++w) s += sS[w * 512 + idx];
    int row = idx >> 5;
    int d = idx & 31;
    ctx[((size_t)(b * SEQ + i0 + row)) * D_MODEL + h * HEAD_DIM + d] = (f16)s;
  }
}

// ---------------------------------------------------------------------------
// LayerNorm(a + bres) * g + beta ; one 256-element row per 256-thread block
// ---------------------------------------------------------------------------
__global__ __launch_bounds__(256) void ln_kernel(
    const float* __restrict__ a, const float* __restrict__ bres,
    const float* __restrict__ g, const float* __restrict__ beta,
    float* __restrict__ outf, f16* __restrict__ outh) {
  __shared__ float red[256];
  const int row = blockIdx.x;
  const int t = threadIdx.x;
  const size_t off = (size_t)row * D_MODEL + t;
  float v = a[off] + bres[off];

  red[t] = v;
  __syncthreads();
  for (int s = 128; s > 0; s >>= 1) {
    if (t < s) red[t] += red[t + s];
    __syncthreads();
  }
  float mu = red[0] * (1.f / D_MODEL);
  __syncthreads();
  float d = v - mu;
  red[t] = d * d;
  __syncthreads();
  for (int s = 128; s > 0; s >>= 1) {
    if (t < s) red[t] += red[t + s];
    __syncthreads();
  }
  float var = red[0] * (1.f / D_MODEL);
  float y = d * __frsqrt_rn(var + 1e-5f) * g[t] + beta[t];
  if (outf) outf[off] = y;
  if (outh) outh[off] = (f16)y;
}

// ---------------------------------------------------------------------------
// Host-side orchestration
// ---------------------------------------------------------------------------
extern "C" void kernel_launch(void* const* d_in, const int* in_sizes, int n_in,
                              void* d_out, int out_size, void* d_ws, size_t ws_size,
                              hipStream_t stream) {
  (void)in_sizes; (void)n_in; (void)out_size; (void)ws_size;

  const float* embed = (const float*)d_in[0];
  // d_in[1] refer_bbox: unused by the reference
  const float* dist  = (const float*)d_in[2];
  const float* qpos  = (const float*)d_in[3];
  const float* in_w  = (const float*)d_in[4];
  const float* in_b  = (const float*)d_in[5];
  const float* out_w = (const float*)d_in[6];
  const float* out_b = (const float*)d_in[7];
  const float* tau_w = (const float*)d_in[8];
  const float* tau_b = (const float*)d_in[9];
  const float* w1    = (const float*)d_in[10];
  const float* b1    = (const float*)d_in[11];
  const float* w2    = (const float*)d_in[12];
  const float* b2    = (const float*)d_in[13];
  const float* g1    = (const float*)d_in[14];
  const float* be1   = (const float*)d_in[15];
  const float* g2    = (const float*)d_in[16];
  const float* be2   = (const float*)d_in[17];

  char* ws = (char*)d_ws;
  const size_t MB = 1ull << 20;
  // Workspace map (peak ~26 MB; h1 and ffn reuse freed regions)
  f16* qk_h   = (f16*)(ws + 0 * MB);             // 2 MB (4096x256)
  f16* emb_h  = (f16*)(ws + 2 * MB);             // 2 MB
  f16* wqkv_h = (f16*)(ws + 4 * MB);             // 768x256 = 384 KB
  f16* wo_h   = (f16*)(ws + 4 * MB + 393216);    // 256x256 = 128 KB
  f16* w1_h   = (f16*)(ws + 4 * MB + 524288);    // 1024x256 = 512 KB
  f16* w2_h   = (f16*)(ws + 4 * MB + 1048576);   // 256x1024 = 512 KB
  float* tauf = (float*)(ws + 4 * MB + 1572864); // 128 KB
  f16* q_h    = (f16*)(ws + 6 * MB);             // 2 MB
  f16* k_h    = (f16*)(ws + 8 * MB);             // 2 MB
  f16* v_h    = (f16*)(ws + 10 * MB);            // 2 MB
  f16* vt_h   = (f16*)(ws + 12 * MB);            // 2 MB
  f16* ctx_h  = (f16*)(ws + 14 * MB);            // 2 MB
  float* tgt_f = (float*)(ws + 16 * MB);         // 4 MB
  float* x_f   = (float*)(ws + 20 * MB);         // 4 MB
  f16* x_h     = (f16*)(ws + 24 * MB);           // 2 MB
  f16* h1_h    = (f16*)(ws + 6 * MB);            // 8 MB (reuse q/k/v/vt)
  float* ffn_f = (float*)(ws + 0 * MB);          // 4 MB (reuse qk/emb)

  const int nelem = ROWS * D_MODEL; // 1M

  // 1. f16 conversions
  cvt_qk_kernel<<<nelem / 256, 256, 0, stream>>>(embed, qpos, qk_h, emb_h, nelem);
  cvt_f16_kernel<<<(3 * D_MODEL * D_MODEL) / 256, 256, 0, stream>>>(in_w, wqkv_h, 3 * D_MODEL * D_MODEL);
  cvt_f16_kernel<<<(D_MODEL * D_MODEL) / 256, 256, 0, stream>>>(out_w, wo_h, D_MODEL * D_MODEL);
  cvt_f16_kernel<<<(D_FFN * D_MODEL) / 256, 256, 0, stream>>>(w1, w1_h, D_FFN * D_MODEL);
  cvt_f16_kernel<<<(D_MODEL * D_FFN) / 256, 256, 0, stream>>>(w2, w2_h, D_MODEL * D_FFN);

  // 2. tau
  tau_kernel<<<(BATCH * SEQ * N_HEADS) / 256, 256, 0, stream>>>(embed, tau_w, tau_b, tauf);

  // 3. Q/K/V projections (WMMA, async-LDS A staging)
  // blocks = (ROWS/16)*NTILES/WAVES
  gemm_wmma_kernel<256, 4, 8><<<128, 256, 0, stream>>>(qk_h, wqkv_h, in_b,
                                                       nullptr, q_h, 0);
  gemm_wmma_kernel<256, 4, 8><<<128, 256, 0, stream>>>(qk_h, wqkv_h + D_MODEL * D_MODEL,
                                                       in_b + D_MODEL, nullptr, k_h, 0);
  gemm_wmma_kernel<256, 4, 8><<<128, 256, 0, stream>>>(emb_h, wqkv_h + 2 * D_MODEL * D_MODEL,
                                                       in_b + 2 * D_MODEL, nullptr, v_h, 0);
  transpose_v_kernel<<<nelem / 256, 256, 0, stream>>>(v_h, vt_h);

  // 4. attention (WMMA QK^T + softmax + WMMA AV)
  attn_kernel<<<BATCH * N_HEADS * (SEQ / 16), 256, 0, stream>>>(q_h, k_h, vt_h,
                                                                tauf, dist, ctx_h);

  // 5. out projection + LN1
  gemm_wmma_kernel<256, 4, 8><<<128, 256, 0, stream>>>(ctx_h, wo_h, out_b,
                                                       tgt_f, nullptr, 0);
  ln_kernel<<<ROWS, 256, 0, stream>>>(embed, tgt_f, g1, be1, x_f, x_h);

  // 6. FFN (WMMA) + LN2 -> d_out
  gemm_wmma_kernel<256, 16, 8><<<512, 256, 0, stream>>>(x_h, w1_h, b1,
                                                        nullptr, h1_h, 1);
  gemm_wmma_kernel<1024, 4, 4><<<256, 128, 0, stream>>>(h1_h, w2_h, b2,
                                                        ffn_f, nullptr, 0);
  ln_kernel<<<ROWS, 256, 0, stream>>>(x_f, ffn_f, g2, be2, (float*)d_out, nullptr);
}
